// self_attention_438086664481
// MI455X (gfx1250) — matrix operand order repaired and checked by reading, not compiled, not run on hardware
//
#include <hip/hip_runtime.h>
#include <hip/hip_bf16.h>
#include <float.h>

// MI455X / gfx1250 fused kernel:
//   qkv-conv(64->32)+BN+PReLU -> G=Q*Q^T (WMMA bf16) -> A=G*Q (WMMA bf16)
//   -> mask+softmax -> proj-conv(32->64)+BN+PReLU -> +residual
// One workgroup per (b,f) pair; everything stays in LDS (256.7KB of 320KB).

typedef __attribute__((ext_vector_type(16))) __bf16 bf16x16;
typedef __attribute__((ext_vector_type(8)))  float  f32x8;

#define B_  4
#define C_  64
#define DC_ 32
#define F_  128
#define T_  512

// ---- WMMA fragment loaders (wave32, 16-bit, contiguous k = 16*(lane/16) + e in both A and B) ----

// A-matrix 16x32 (MxK), row-major source with leading dim `ld`.
// lane<16: row M=lane, K = 0..15
// lane>=16: row M=lane-16, K = 16..31
__device__ __forceinline__ bf16x16 frag_a_rowmajor(const __bf16* m, int ld,
                                                   int mBase, int kBase, int lane) {
  const int half = lane >> 4, lr = lane & 15;
  const __bf16* row = m + (mBase + lr) * ld + kBase + half * 16;
  bf16x16 a;
#pragma unroll
  for (int e = 0; e < 8; ++e) a[e] = row[e];
#pragma unroll
  for (int e = 0; e < 8; ++e) a[8 + e] = row[8 + e];
  return a;
}

// B-matrix 32x16 (KxN) where B[k][n] = src[(nBase+n)*ld + kBase + k]
// (i.e. B is the transpose of a row-major tile: used for Q^T in G = Q*Q^T).
// lane<16: col N=lane, K=0..15 ; lane>=16: col N=lane-16, K=16..31
__device__ __forceinline__ bf16x16 frag_b_transposed(const __bf16* m, int ld,
                                                     int nBase, int kBase, int lane) {
  const int half = lane >> 4, lr = lane & 15;
  const __bf16* row = m + (nBase + lr) * ld + kBase + half * 16;
  bf16x16 b;
#pragma unroll
  for (int e = 0; e < 16; ++e) b[e] = row[e];
  return b;
}

// B-matrix 32x16 (KxN) where B[k][n] = src[(kBase+k)*ld + nBase + n] (row-major KxN)
__device__ __forceinline__ bf16x16 frag_b_rowmajor(const __bf16* m, int ld,
                                                   int kBase, int nBase, int lane) {
  const int half = lane >> 4, lr = lane & 15;
  const __bf16* col = m + (kBase + half * 16) * ld + nBase + lr;
  bf16x16 b;
#pragma unroll
  for (int e = 0; e < 16; ++e) b[e] = col[e * ld];
  return b;
}

__global__ __launch_bounds__(256)
void attn_fused_gfx1250(const float* __restrict__ inp,
                        const float* __restrict__ w1, const float* __restrict__ b1,
                        const float* __restrict__ g1, const float* __restrict__ be1,
                        const float* __restrict__ m1, const float* __restrict__ v1,
                        const float* __restrict__ a1,
                        const float* __restrict__ wp, const float* __restrict__ bp,
                        const float* __restrict__ g2, const float* __restrict__ be2,
                        const float* __restrict__ m2, const float* __restrict__ v2,
                        const float* __restrict__ a2,
                        float* __restrict__ out)
{
  extern __shared__ char smem[];
  float*  X   = (float*)(smem);                 // [64][512] fp32 input tile (also residual)
  __bf16* Qbf = (__bf16*)(smem + 131072);       // [32][512] bf16 qkv
  float*  Gp  = (float*)(smem + 163840);        // [2][32][32] fp32 partial G
  __bf16* Gbf = (__bf16*)(smem + 172032);       // [32][32] bf16 (scaled) G
  float*  P   = (float*)(smem + 174080);        // [32][512] fp32 scores -> probs
  float*  w1f = (float*)(smem + 239616);        // [32][64] BN-folded W1
  float*  w2f = (float*)(smem + 247808);        // [64][32] BN-folded Wp
  float*  b1f = (float*)(smem + 256000);        // [32]
  float*  b2f = (float*)(smem + 256128);        // [64]
  float*  sc1 = (float*)(smem + 256384);        // [32]
  float*  sc2 = (float*)(smem + 256512);        // [64]

  const int tid  = threadIdx.x;
  const int wave = tid >> 5;
  const int lane = tid & 31;
  const int b    = blockIdx.x >> 7;             // 512 blocks = B*F
  const int f    = blockIdx.x & (F_ - 1);
  const float alpha1 = a1[0], alpha2 = a2[0];

  // ---- phase 0: fold eval-BatchNorm into weights; stage X = inp[b,:,f,:] ----
  if (tid < DC_) {
    float s = g1[tid] * rsqrtf(v1[tid] + 1e-5f);
    sc1[tid] = s;
    b1f[tid] = (b1[tid] - m1[tid]) * s + be1[tid];
  }
  if (tid < C_) {
    float s = g2[tid] * rsqrtf(v2[tid] + 1e-5f);
    sc2[tid] = s;
    b2f[tid] = (bp[tid] - m2[tid]) * s + be2[tid];
  }
  __syncthreads();
  for (int i = tid; i < DC_ * C_; i += 256) w1f[i] = w1[i] * sc1[i >> 6];
  for (int i = tid; i < C_ * DC_; i += 256) w2f[i] = wp[i] * sc2[i >> 5];

  const float4* inp4 = (const float4*)inp;
  for (int i = tid; i < C_ * (T_ / 4); i += 256) {
    int k = i >> 7, t4 = i & 127;                      // 128 float4 per row
    ((float4*)X)[k * (T_ / 4) + t4] =
        inp4[((size_t)((b * C_ + k) * F_ + f) * T_ >> 2) + t4];
  }
  __syncthreads();

  // ---- phase 1: Q = PReLU(BN(W1 X)) in fp32, stored bf16 for the matrix cores ----
  for (int i = tid; i < DC_ * T_; i += 256) {
    int c = i >> 9, t = i & (T_ - 1);
    const float* wrow = w1f + c * C_;
    float acc = b1f[c];
#pragma unroll 8
    for (int k = 0; k < C_; ++k) acc = fmaf(wrow[k], X[k * T_ + t], acc);
    acc = acc >= 0.f ? acc : alpha1 * acc;
    Qbf[c * T_ + t] = (__bf16)acc;
  }
  __syncthreads();

  // ---- phase 2: G = Q * Q^T  (32x32, K=512) via v_wmma_f32_16x16x32_bf16 ----
  // 8 waves = 2 M-tiles x 2 N-tiles x 2 K-halves
  {
    const int mi = wave & 1, ni = (wave >> 1) & 1, kh = wave >> 2;
    f32x8 acc = {};
#pragma unroll
    for (int s = 0; s < 8; ++s) {
      const int t0 = kh * 256 + s * 32;
      bf16x16 fa = frag_a_rowmajor(Qbf, T_, mi * 16, t0, lane);   // A[m,k]=Q[m,t0+k]
      bf16x16 fb = frag_b_transposed(Qbf, T_, ni * 16, t0, lane); // B[k,n]=Q[n,t0+k]
      acc = __builtin_amdgcn_wmma_f32_16x16x32_bf16(false, fa, false, fb,
                                                    (short)0, acc, false, false);
    }
    const int half = lane >> 4, lr = lane & 15;
#pragma unroll
    for (int r = 0; r < 8; ++r)
      Gp[kh * 1024 + (mi * 16 + half * 8 + r) * 32 + ni * 16 + lr] = acc[r];
  }
  __syncthreads();
  // combine K-halves, fold the 1/sqrt(d_c) attention scale, convert to bf16
  for (int i = tid; i < 1024; i += 256)
    Gbf[i] = (__bf16)((Gp[i] + Gp[1024 + i]) * 0.17677669529663687f);
  __syncthreads();

  // ---- phase 3: A = G * Q  (32x512, K=32 -> one WMMA per tile) ----
  {
    const int half = lane >> 4, lr = lane & 15;
#pragma unroll
    for (int mi = 0; mi < 2; ++mi) {
      bf16x16 fa = frag_a_rowmajor(Gbf, 32, mi * 16, 0, lane);
#pragma unroll
      for (int it = 0; it < 4; ++it) {
        const int nj = wave + it * 8;                            // 32 N-tiles / 8 waves
        bf16x16 fb = frag_b_rowmajor(Qbf, T_, 0, nj * 16, lane); // B[k,n]=Q[k,nj*16+n]
        f32x8 d = {};
        d = __builtin_amdgcn_wmma_f32_16x16x32_bf16(false, fa, false, fb,
                                                    (short)0, d, false, false);
#pragma unroll
        for (int r = 0; r < 8; ++r)
          P[(mi * 16 + half * 8 + r) * T_ + nj * 16 + lr] = d[r];
      }
    }
  }
  __syncthreads();

  // ---- phase 4: causal mask (t - f >= 385 -> -FLT_MAX) + softmax over T ----
  {
    const int c = tid >> 3, sub = tid & 7;        // 8 lanes per row, same wave
    float* row = P + c * T_;
    const int tlim = (f + 385 > T_) ? T_ : f + 385;
    const int t0 = sub * 64;
    float mx = -FLT_MAX;
    for (int t = t0; t < t0 + 64; ++t) if (t < tlim) mx = fmaxf(mx, row[t]);
#pragma unroll
    for (int m = 1; m < 8; m <<= 1) mx = fmaxf(mx, __shfl_xor(mx, m, 32));
    float sum = 0.f;
    for (int t = t0; t < t0 + 64; ++t) if (t < tlim) sum += __expf(row[t] - mx);
#pragma unroll
    for (int m = 1; m < 8; m <<= 1) sum += __shfl_xor(sum, m, 32);
    const float inv = 1.0f / sum;
    for (int t = t0; t < t0 + 64; ++t)
      row[t] = (t < tlim) ? __expf(row[t] - mx) * inv : 0.f;
  }
  __syncthreads();

  // ---- phase 5: out = PReLU(BN(Wp P)) + X, vectorized float4 stores ----
  float4* out4 = (float4*)out;
  for (int i = tid; i < C_ * (T_ / 4); i += 256) {
    const int co = i >> 7, t4 = (i & 127) << 2;
    const float* wrow = w2f + co * DC_;
    float acc0 = b2f[co], acc1 = acc0, acc2 = acc0, acc3 = acc0;
#pragma unroll 8
    for (int j = 0; j < DC_; ++j) {
      const float w = wrow[j];
      const float* prow = P + j * T_ + t4;
      acc0 = fmaf(w, prow[0], acc0);
      acc1 = fmaf(w, prow[1], acc1);
      acc2 = fmaf(w, prow[2], acc2);
      acc3 = fmaf(w, prow[3], acc3);
    }
    const float* xr = X + co * T_ + t4;
    float4 o;
    o.x = (acc0 >= 0.f ? acc0 : alpha2 * acc0) + xr[0];
    o.y = (acc1 >= 0.f ? acc1 : alpha2 * acc1) + xr[1];
    o.z = (acc2 >= 0.f ? acc2 : alpha2 * acc2) + xr[2];
    o.w = (acc3 >= 0.f ? acc3 : alpha2 * acc3) + xr[3];
    out4[((size_t)((b * C_ + co) * F_ + f) * T_ >> 2) + (t4 >> 2)] = o;
  }
}

extern "C" void kernel_launch(void* const* d_in, const int* in_sizes, int n_in,
                              void* d_out, int out_size, void* d_ws, size_t ws_size,
                              hipStream_t stream) {
  const float* inp = (const float*)d_in[0];
  const float* w1  = (const float*)d_in[1];
  const float* b1  = (const float*)d_in[2];
  const float* g1  = (const float*)d_in[3];
  const float* be1 = (const float*)d_in[4];
  const float* m1  = (const float*)d_in[5];
  const float* v1  = (const float*)d_in[6];
  const float* a1  = (const float*)d_in[7];
  const float* wp  = (const float*)d_in[8];
  const float* bp  = (const float*)d_in[9];
  const float* g2  = (const float*)d_in[10];
  const float* be2 = (const float*)d_in[11];
  const float* m2  = (const float*)d_in[12];
  const float* v2  = (const float*)d_in[13];
  const float* a2  = (const float*)d_in[14];

  const size_t smemBytes = 256768;   // 256.7KB of gfx1250's 320KB/WGP LDS
  attn_fused_gfx1250<<<dim3(B_ * F_), dim3(256), smemBytes, stream>>>(
      inp, w1, b1, g1, be1, m1, v1, a1, wp, bp, g2, be2, m2, v2, a2,
      (float*)d_out);
}
